// SE3TransformerIterative_41455024341539
// MI455X (gfx1250) — matrix operand order repaired and checked by reading, not compiled, not run on hardware
//
// SE(3)-Transformer iterative forward for MI455X (gfx1250), compile-only target.
//
// Design notes (MI455X reasoning):
//  * wave32: K=32 edges per dst node == exactly one wave -> softmax and
//    attention aggregation are pure shfl_xor wave reductions, no cross-wave sync.
//  * The only large dense GEMM is the radial MLP second layer:
//    [E=131072, 32] @ [32, 80]  -> tiled as V_WMMA_F32_16X16X32_F16 (f16 A/B,
//    f32 accum; K=32 dot is safe in f16 inputs). A-tile comes from LDS
//    (phase-1 MLP written as f16), per ISA 7.12.2 16-bit A 16x32 layout;
//    B-tile per the 32x16 B layout (K across dwords, N across lanes).
//  * Everything else is latency-bound (10 GFLOP, ~0.5GB traffic vs 23.3TB/s),
//    so we minimize kernel count and keep edge data dst-major.
//
// Input flattening assumption: setup_inputs() order x, w, ones, params;
// params pytree flattened depth-first with dict keys sorted (JAX convention):
//   per attn layer: Wk_0l,[Wk_l0],Wk_same,Wout{'0'..},Wq,Wr1,Wr2,Wv_0l,[Wv_l0],
//   Wv_same,br1,br2 ; then norm[0..5].  => 107 leaves total.
//
// Workspace: ~45 MB carved from d_ws.

#include <hip/hip_runtime.h>

typedef __attribute__((ext_vector_type(16))) _Float16 v16h;
typedef __attribute__((ext_vector_type(8)))  float    v8f;

#define NNODES_PER_B 512
#define KK   32
#define BN   4096
#define NE   131072
#define EPSF 1e-8f

struct LP {
  const float* Wk_0l; const float* Wk_l0; const float* Wk_same;
  const float* Wout0; const float* Wout1; const float* Wout2; const float* Wout3;
  const float* Wq; const float* Wr1; const float* Wr2;
  const float* Wv_0l; const float* Wv_l0; const float* Wv_same;
  const float* br1; const float* br2; const float* normb;
};

// degree helpers: m-offsets (cum 2l+1) and f-buffer offsets (16 ch per degree)
__device__ __forceinline__ int moffd(int l){ return (l==0)?0:((l==1)?1:((l==2)?4:9)); }
__device__ __forceinline__ int foffd(int l){ return (l==0)?0:((l==1)?16:((l==2)?64:144)); }
__device__ __forceinline__ float Yval(int l, int m, const float* y){
  return (l==1) ? y[m] : ((l==2) ? y[3+m] : y[8+m]);  // l==0 never queried
}
// CFG 0: layer0 (din={0:1}, same={0}, up={1,2,3}, cv=4, npaths=4)
// CFG 1: mid    (din=all16, same={0..3}, up={1,2,3}, down={1,2,3}, cv=4, npaths=10)
// CFG 2: final  (same={1}, up={1}, cv=1, npaths=2, dout={1}, cout=1)
template<int CFG> __device__ __forceinline__ bool in_same(int l){ return CFG==1 ? true : (CFG==0 ? (l==0) : (l==1)); }
template<int CFG> __device__ __forceinline__ bool in_up(int l){ return (CFG==2) ? (l==1) : (l>=1); }
template<int CFG> __device__ __forceinline__ int pi_same(int l){ return (CFG==1)?l:0; }
template<int CFG> __device__ __forceinline__ int pi_up(int l){ return (CFG==0)?l:((CFG==1)?(3+l):1); }
template<int CFG> __device__ __forceinline__ int upidx(int l){ return (CFG==2)?0:(l-1); }

// ---------------------------------------------------------------- top-k edges
__global__ void k_topk(const float* __restrict__ w, int* __restrict__ src,
                       float* __restrict__ wkb){
  int node = blockIdx.x, lane = threadIdx.x;
  int b = node >> 9, dl = node & 511;
  const float* row = w + (size_t)node * (NNODES_PER_B - 1);
  float vals[16];
  #pragma unroll
  for (int t = 0; t < 16; t++){ int j = lane + t*32; vals[t] = (j < NNODES_PER_B-1) ? row[j] : -3.0e38f; }
  for (int kk = 0; kk < KK; kk++){
    float bv = -3.0e38f; int bj = 0x7fffffff;
    #pragma unroll
    for (int t = 0; t < 16; t++){ int j = lane + t*32; if (vals[t] > bv){ bv = vals[t]; bj = j; } }
    #pragma unroll
    for (int off = 16; off >= 1; off >>= 1){
      float ov = __shfl_xor(bv, off, 32); int oj = __shfl_xor(bj, off, 32);
      if (ov > bv || (ov == bv && oj < bj)){ bv = ov; bj = oj; }
    }
    if ((bj & 31) == lane) vals[bj >> 5] = -3.0e38f;
    if (lane == 0){
      int e = node*KK + kk;
      int sl = (bj < dl) ? bj : bj + 1;
      src[e] = b*NNODES_PER_B + sl;
      wkb[e] = bv;
    }
  }
}

// ------------------------------------------------------------------- init
__global__ void k_init(const float* __restrict__ x, const float* __restrict__ ones,
                       float* __restrict__ pos, float* __restrict__ fA){
  int i = blockIdx.x*blockDim.x + threadIdx.x;
  if (i < BN*3) pos[i] = x[i];
  if (i < BN)   fA[(size_t)i*256] = ones[i];   // f0 channel 0; rest unread at layer0
}

// ------------------------------------------------------------- edge geometry
__global__ void k_geom(const int* __restrict__ src, const float* __restrict__ pos,
                       float* __restrict__ Y, float* __restrict__ rb){
  int e = blockIdx.x*blockDim.x + threadIdx.x;
  if (e >= NE) return;
  int d = e >> 5, s = src[e];
  float rx = pos[s*3+0]-pos[d*3+0], ry = pos[s*3+1]-pos[d*3+1], rz = pos[s*3+2]-pos[d*3+2];
  float r = sqrtf(rx*rx + ry*ry + rz*rz + EPSF);
  rb[e] = r;
  float inv = 1.0f/r, x = rx*inv, y = ry*inv, z = rz*inv;
  float x2 = x*x, y2 = y*y, z2 = z*z;
  float* Ye = Y + (size_t)e*15;
  Ye[0]  = 0.4886025119029199f*y;  Ye[1] = 0.4886025119029199f*z;  Ye[2] = 0.4886025119029199f*x;
  Ye[3]  = 1.0925484305920792f*x*y; Ye[4] = 1.0925484305920792f*y*z;
  Ye[5]  = 0.31539156525252005f*(3.f*z2 - 1.f);
  Ye[6]  = 1.0925484305920792f*x*z; Ye[7] = 0.5462742152960396f*(x2 - y2);
  Ye[8]  = 0.5900435899266435f*y*(3.f*x2 - y2); Ye[9] = 2.890611442640554f*x*y*z;
  Ye[10] = 0.4570457994644658f*y*(5.f*z2 - 1.f); Ye[11] = 0.3731763325901154f*z*(5.f*z2 - 3.f);
  Ye[12] = 0.4570457994644658f*x*(5.f*z2 - 1.f); Ye[13] = 1.445305721320277f*z*(x2 - y2);
  Ye[14] = 0.5900435899266435f*x*(x2 - 3.f*y2);
}

// -------------------------------------------------- radial MLP (WMMA GEMM)
// h = relu(rad_in @ Wr1 + br1) [64 edges x 32] in LDS (f16), then
// R = h @ Wr2 + br2  via v_wmma_f32_16x16x32_f16, stored f16.
template<int CFG>
__global__ void k_radial(const float* __restrict__ rb, const float* __restrict__ wkb,
                         LP lp, _Float16* __restrict__ R){
  constexpr int NP2 = (CFG==0) ? 32 : ((CFG==1) ? 80 : 4);
  constexpr int NT  = (CFG==0) ? 2  : ((CFG==1) ? 5  : 1);
  __shared__ _Float16 hsm[64][40];
  int tid = threadIdx.x;          // 128 threads = 4 waves, 64 edges/block
  int e0 = blockIdx.x * 64;
  { // phase 1: 2 threads per edge, 16 hidden each
    int s = tid >> 1, kb = (tid & 1) * 16;
    float r = rb[e0 + s], wv = wkb[e0 + s];
    for (int j = 0; j < 16; j++){
      int k = kb + j;
      float h = lp.Wr1[k]*r + lp.Wr1[32 + k]*wv + lp.br1[k];
      hsm[s][k] = (_Float16)fmaxf(h, 0.0f);
    }
  }
  __syncthreads();
  int wv_ = tid >> 5, L = tid & 31;
  int M16 = L & 15, hi = L >> 4;
  v16h a;                          // ISA 7.12.2: 16-bit A 16x32 layout
  #pragma unroll
  for (int jj = 0; jj < 16; jj++){
    int i = jj >> 1;
    int kidx = ((i < 4) ? 2*i : 8 + 2*i) + hi*8 + (jj & 1);
    a[jj] = hsm[wv_*16 + M16][kidx];
  }
  #pragma unroll
  for (int nt = 0; nt < NT; nt++){
    int ncol = nt*16 + M16;
    v16h b;                        // 16-bit B 32x16: K across dwords, N=lane
    #pragma unroll
    for (int jj = 0; jj < 16; jj++){
      int k = (jj < 8) ? (hi*8 + jj) : (16 + hi*8 + (jj - 8));
      b[jj] = (ncol < NP2) ? (_Float16)lp.Wr2[k*NP2 + ncol] : (_Float16)0.0f;
    }
    v8f c = {};
    c = __builtin_amdgcn_wmma_f32_16x16x32_f16(false, a, false, b, (short)0, c, false, false);
    if (ncol < NP2){
      float bias = lp.br2[ncol];
      #pragma unroll
      for (int rr = 0; rr < 8; rr++){
        int m = rr + hi*8;
        R[(size_t)(e0 + wv_*16 + m)*NP2 + ncol] = (_Float16)(c[rr] + bias);
      }
    }
  }
}

// -------------------------------------------- per-node self-interactions
// node buffer (stride 352 f32): uv[0..64) uk[64..128) q[128..192)
//   dv[192..252) dk[252..312) sv[312..324) sk[324..336)
template<int CFG>
__global__ void k_node(const float* __restrict__ f, LP lp, float* __restrict__ nb){
  constexpr int CV  = (CFG==2) ? 1 : 4;
  constexpr int ND  = (CFG==1) ? 3 : 0;
  constexpr int NU  = (CFG==2) ? 1 : 3;
  constexpr int CIN = (CFG==0) ? 1 : 16;
  int n = blockIdx.x;
  const float* fn = f + (size_t)n*256;
  float* out = nb + (size_t)n*352;
  for (int idx = threadIdx.x; idx < 336; idx += blockDim.x){
    if (idx < 192){
      int r3 = idx >> 6, k = idx & 63, j = k >> 2, c = k & 3;
      int l, m;
      if (j < 1){ l=0; m=j; } else if (j < 4){ l=1; m=j-1; }
      else if (j < 9){ l=2; m=j-4; } else { l=3; m=j-9; }
      bool act = (CFG==1) || (CFG==0 && l==0) || (CFG==2 && l==1 && c==0);
      if (!act) continue;
      const float* W = (r3==0) ? lp.Wv_same : ((r3==1) ? lp.Wk_same : lp.Wq);
      int il = (CFG==1) ? l : 0;
      int ml = 2*l + 1;
      const float* Wr = W + (il*CV + c)*CIN;
      const float* fr = fn + foffd(l) + m;
      float acc = 0.f;
      for (int ci = 0; ci < CIN; ci++) acc += Wr[ci] * fr[ci*ml];
      out[idx] = acc;
    } else if (idx < 312){
      if (ND == 0) continue;
      int isK = (idx >= 252);
      int k = idx - (isK ? 252 : 192), j = k >> 2, c = k & 3;
      int l, m;
      if (j < 3){ l=1; m=j; } else if (j < 8){ l=2; m=j-3; } else { l=3; m=j-8; }
      const float* W = isK ? lp.Wk_l0 : lp.Wv_l0;
      int ml = 2*l + 1;
      const float* Wr = W + ((l-1)*4 + c)*16;
      const float* fr = fn + foffd(l) + m;
      float acc = 0.f;
      for (int ci = 0; ci < 16; ci++) acc += Wr[ci] * fr[ci*ml];
      out[idx] = acc;
    } else {
      int isK = (idx >= 324);
      int u = idx - (isK ? 324 : 312), ui = u >> 2, c = u & 3;
      if (ui >= NU || c >= CV) continue;
      const float* W = isK ? lp.Wk_0l : lp.Wv_0l;
      const float* Wr = W + (ui*CV + c)*CIN;
      float acc = 0.f;
      for (int ci = 0; ci < CIN; ci++) acc += Wr[ci] * fn[ci];  // f0
      out[idx] = acc;
    }
  }
}

// -------------------- attention: logit, wave softmax, aggregation, Wout+norm
template<int CFG>
__global__ void k_attn(const int* __restrict__ src, const _Float16* __restrict__ R,
                       const float* __restrict__ Y, const float* __restrict__ nb,
                       const float* __restrict__ fin, LP lp,
                       float* __restrict__ fout, float* __restrict__ upd){
  constexpr int NP  = (CFG==0) ? 4 : ((CFG==1) ? 10 : 2);
  constexpr int CV  = (CFG==2) ? 1 : 4;
  constexpr int NU  = (CFG==2) ? 1 : 3;
  constexpr int NP2 = 2*NP*CV;
  constexpr float RSD = (CFG==0) ? 0.5f : ((CFG==1) ? 0.125f : 0.5773502691896258f);
  __shared__ float agg[64];
  __shared__ float tmp[256];
  int n = blockIdx.x, lane = threadIdx.x;
  int e = n*KK + lane;
  int s = src[e];
  const _Float16* Re = R + (size_t)e*NP2;
  float Rv[NP][CV], Rk[NP][CV];
  #pragma unroll
  for (int p = 0; p < NP; p++)
    #pragma unroll
    for (int c = 0; c < CV; c++){
      Rv[p][c] = (float)Re[p*CV + c];
      Rk[p][c] = (float)Re[NP*CV + p*CV + c];
    }
  float y[15];
  #pragma unroll
  for (int i = 0; i < 15; i++) y[i] = Y[(size_t)e*15 + i];
  const float* nbs = nb + (size_t)s*352;
  const float* nbd = nb + (size_t)n*352;
  float sv[NU][CV], skv[NU][CV];
  #pragma unroll
  for (int u = 0; u < NU; u++)
    #pragma unroll
    for (int c = 0; c < CV; c++){ sv[u][c] = nbs[312 + u*4 + c]; skv[u][c] = nbs[324 + u*4 + c]; }
  float vdown[CV] = {}, kdown[CV] = {};
  if (CFG == 1){
    #pragma unroll
    for (int ld = 1; ld < 4; ld++){
      int dm = (ld==1) ? 0 : ((ld==2) ? 3 : 8);
      int ml = 2*ld + 1;
      #pragma unroll
      for (int c = 0; c < CV; c++){
        float dv = 0.f, dk = 0.f;
        #pragma unroll
        for (int m = 0; m < 7; m++){
          if (m >= ml) break;
          float yy = y[dm + m];
          dv += nbs[192 + (dm + m)*4 + c]*yy;
          dk += nbs[252 + (dm + m)*4 + c]*yy;
        }
        vdown[c] += Rv[6 + ld][c]*dv;
        kdown[c] += Rk[6 + ld][c]*dk;
      }
    }
  }
  // ---- logit over same paths
  float logit = 0.f;
  #pragma unroll
  for (int l = 0; l < 4; l++){
    if (!in_same<CFG>(l)) continue;
    int ml = 2*l + 1;
    #pragma unroll
    for (int c = 0; c < CV; c++)
      #pragma unroll
      for (int m = 0; m < 7; m++){
        if (m >= ml) break;
        int ix = (moffd(l) + m)*4 + c;
        float kv = Rk[pi_same<CFG>(l)][c] * nbs[64 + ix];
        if (l > 0 && in_up<CFG>(l)) kv += Rk[pi_up<CFG>(l)][c] * skv[upidx<CFG>(l)][c] * Yval(l, m, y);
        if (CFG == 1 && l == 0) kv += kdown[c];
        logit += nbd[128 + ix] * kv;
      }
  }
  logit *= RSD;
  // ---- wave32 softmax over K=32 edges of this dst
  float mx = logit;
  #pragma unroll
  for (int off = 16; off >= 1; off >>= 1) mx = fmaxf(mx, __shfl_xor(mx, off, 32));
  float ex = __expf(logit - mx);
  float sm = ex;
  #pragma unroll
  for (int off = 16; off >= 1; off >>= 1) sm += __shfl_xor(sm, off, 32);
  float alpha = ex / sm;
  // ---- aggregation (wave reduce per component)
  #pragma unroll
  for (int l = 0; l < 4; l++){
    if (CFG == 2 && l != 1) continue;
    int ml = 2*l + 1;
    #pragma unroll
    for (int c = 0; c < CV; c++)
      #pragma unroll
      for (int m = 0; m < 7; m++){
        if (m >= ml) break;
        int ix = (moffd(l) + m)*4 + c;
        float vv = 0.f;
        if (in_same<CFG>(l)) vv += Rv[pi_same<CFG>(l)][c] * nbs[ix];
        if (l > 0 && in_up<CFG>(l)) vv += Rv[pi_up<CFG>(l)][c] * sv[upidx<CFG>(l)][c] * Yval(l, m, y);
        if (CFG == 1 && l == 0) vv += vdown[c];
        float acc = alpha * vv;
        #pragma unroll
        for (int off = 16; off >= 1; off >>= 1) acc += __shfl_xor(acc, off, 32);
        if (lane == 0) agg[ix] = acc;
      }
  }
  __syncthreads();
  if (CFG != 2){
    // ---- Wout: out[l][o][m] = Wout_l @ cat(agg, f_in)
    #pragma unroll
    for (int t = 0; t < 8; t++){
      int idx = lane + t*32;
      int j = idx >> 4, o = idx & 15;
      int l, m;
      if (j < 1){ l=0; m=0; } else if (j < 4){ l=1; m=j-1; }
      else if (j < 9){ l=2; m=j-4; } else { l=3; m=j-9; }
      const float* W = (l==0) ? lp.Wout0 : ((l==1) ? lp.Wout1 : ((l==2) ? lp.Wout2 : lp.Wout3));
      int dinch = (CFG==0) ? ((l==0) ? 1 : 0) : 16;
      int ccat = 4 + dinch;
      int ml = 2*l + 1;
      float v = 0.f;
      for (int c = 0; c < 4; c++) v += W[o*ccat + c]*agg[(moffd(l) + m)*4 + c];
      for (int ci = 0; ci < dinch; ci++)
        v += W[o*ccat + 4 + ci]*fin[(size_t)n*256 + foffd(l) + ci*ml + m];
      tmp[foffd(l) + o*ml + m] = v;
    }
    __syncthreads();
    // ---- fused GNormBias: relu(||f||+b) * f/||f||
    #pragma unroll
    for (int t = 0; t < 2; t++){
      int p = lane + t*32;
      int l = p >> 4, o = p & 15, ml = 2*l + 1;
      float s2 = 0.f;
      for (int m = 0; m < ml; m++){ float vv = tmp[foffd(l) + o*ml + m]; s2 += vv*vv; }
      float nn = sqrtf(s2 + EPSF);
      float scale = fmaxf(nn + lp.normb[l*16 + o], 0.f) / nn;
      for (int m = 0; m < ml; m++)
        fout[(size_t)n*256 + foffd(l) + o*ml + m] = tmp[foffd(l) + o*ml + m]*scale;
    }
  } else {
    // final layer: dout={1}, cout=1 -> position update directly
    if (lane < 3){
      int m = lane;
      float v = lp.Wout1[0]*agg[(1 + m)*4];
      for (int ci = 0; ci < 16; ci++)
        v += lp.Wout1[1 + ci]*fin[(size_t)n*256 + 16 + ci*3 + m];
      upd[n*3 + m] = v;
    }
  }
}

// ------------------------------------------------------ position update
__global__ void k_posupd(const float* __restrict__ fcur, const float* __restrict__ updb,
                         int it, float* __restrict__ pos, float* __restrict__ dout){
  int n = blockIdx.x*blockDim.x + threadIdx.x;
  if (n >= BN) return;
  float u0, u1, u2;
  if (it < 2){
    u0 = fcur[(size_t)n*256 + 16]; u1 = fcur[(size_t)n*256 + 17]; u2 = fcur[(size_t)n*256 + 18];
  } else {
    u0 = updb[n*3]; u1 = updb[n*3 + 1]; u2 = updb[n*3 + 2];
  }
  pos[n*3 + 0] += u0; pos[n*3 + 1] += u1; pos[n*3 + 2] += u2;
  dout[12288 + it*4096 + n] = sqrtf(u0*u0 + u1*u1 + u2*u2);
}

__global__ void k_final(const float* __restrict__ pos, const float* __restrict__ x,
                        float* __restrict__ dout){
  int n = blockIdx.x*blockDim.x + threadIdx.x;
  if (n >= BN) return;
  float p0 = pos[n*3], p1 = pos[n*3 + 1], p2 = pos[n*3 + 2];
  float d0 = p0 - x[n*3], d1 = p1 - x[n*3 + 1], d2 = p2 - x[n*3 + 2];
  dout[n*3 + 0] = p0; dout[n*3 + 1] = p1; dout[n*3 + 2] = p2;
  dout[24576 + n] = sqrtf(d0*d0 + d1*d1 + d2*d2);
}

// ------------------------------------------------------------------ host
extern "C" void kernel_launch(void* const* d_in, const int* in_sizes, int n_in,
                              void* d_out, int out_size, void* d_ws, size_t ws_size,
                              hipStream_t stream){
  (void)in_sizes; (void)n_in; (void)out_size; (void)ws_size;
  const float* x    = (const float*)d_in[0];
  const float* w    = (const float*)d_in[1];
  const float* ones = (const float*)d_in[2];

  LP lps[7];
  int idx = 3;
  for (int li = 0; li < 7; li++){
    LP p; p.Wk_l0 = nullptr; p.Wv_l0 = nullptr; p.Wout0 = p.Wout2 = p.Wout3 = nullptr; p.normb = nullptr;
    bool mid = (li >= 1 && li <= 5);
    p.Wk_0l = (const float*)d_in[idx++];
    if (mid) p.Wk_l0 = (const float*)d_in[idx++];
    p.Wk_same = (const float*)d_in[idx++];
    if (li == 6){ p.Wout1 = (const float*)d_in[idx++]; }
    else {
      p.Wout0 = (const float*)d_in[idx++]; p.Wout1 = (const float*)d_in[idx++];
      p.Wout2 = (const float*)d_in[idx++]; p.Wout3 = (const float*)d_in[idx++];
    }
    p.Wq  = (const float*)d_in[idx++];
    p.Wr1 = (const float*)d_in[idx++];
    p.Wr2 = (const float*)d_in[idx++];
    p.Wv_0l = (const float*)d_in[idx++];
    if (mid) p.Wv_l0 = (const float*)d_in[idx++];
    p.Wv_same = (const float*)d_in[idx++];
    p.br1 = (const float*)d_in[idx++];
    p.br2 = (const float*)d_in[idx++];
    lps[li] = p;
  }
  for (int i = 0; i < 6; i++) lps[i].normb = (const float*)d_in[idx++];

  // workspace carve (~45 MB)
  unsigned char* base = (unsigned char*)d_ws;
  size_t off = 0;
  auto carve = [&](size_t bytes) -> void* {
    void* p = base + off; off += (bytes + 255) & ~(size_t)255; return p;
  };
  float*    pos  = (float*)carve((size_t)BN*3*sizeof(float));
  int*      srcb = (int*)carve((size_t)NE*sizeof(int));
  float*    wkb  = (float*)carve((size_t)NE*sizeof(float));
  float*    rb   = (float*)carve((size_t)NE*sizeof(float));
  float*    Yb   = (float*)carve((size_t)NE*15*sizeof(float));
  _Float16* Rb   = (_Float16*)carve((size_t)NE*80*sizeof(_Float16));
  float*    fA   = (float*)carve((size_t)BN*256*sizeof(float));
  float*    fB   = (float*)carve((size_t)BN*256*sizeof(float));
  float*    nbuf = (float*)carve((size_t)BN*352*sizeof(float));
  float*    updb = (float*)carve((size_t)BN*3*sizeof(float));
  float*    dout = (float*)d_out;

  k_topk<<<BN, 32, 0, stream>>>(w, srcb, wkb);
  k_init<<<48, 256, 0, stream>>>(x, ones, pos, fA);

  float* fin = fA; float* fout = fB;
  for (int it = 0; it < 3; it++){
    k_geom<<<NE/256, 256, 0, stream>>>(srcb, pos, Yb, rb);
    for (int lr = 0; lr < 2; lr++){
      int li = it*2 + lr;
      if (li == 0){
        k_radial<0><<<NE/64, 128, 0, stream>>>(rb, wkb, lps[0], Rb);
        k_node<0><<<BN, 128, 0, stream>>>(fin, lps[0], nbuf);
        k_attn<0><<<BN, 32, 0, stream>>>(srcb, Rb, Yb, nbuf, fin, lps[0], fout, updb);
      } else {
        k_radial<1><<<NE/64, 128, 0, stream>>>(rb, wkb, lps[li], Rb);
        k_node<1><<<BN, 128, 0, stream>>>(fin, lps[li], nbuf);
        k_attn<1><<<BN, 32, 0, stream>>>(srcb, Rb, Yb, nbuf, fin, lps[li], fout, updb);
      }
      float* t = fin; fin = fout; fout = t;
    }
    if (it == 2){
      k_radial<2><<<NE/64, 128, 0, stream>>>(rb, wkb, lps[6], Rb);
      k_node<2><<<BN, 128, 0, stream>>>(fin, lps[6], nbuf);
      k_attn<2><<<BN, 32, 0, stream>>>(srcb, Rb, Yb, nbuf, fin, lps[6], fout, updb);
    }
    k_posupd<<<16, 256, 0, stream>>>(fin, updb, it, pos, dout);
  }
  k_final<<<16, 256, 0, stream>>>(pos, x, dout);
}
